// Sequence2Sequence_85985245266066
// MI455X (gfx1250) — matrix operand order repaired and checked by reading, not compiled, not run on hardware
//
#include <hip/hip_runtime.h>
#include <cstdint>
#include <cstddef>

// Problem dims (match reference)
#define BB  64
#define SS  1024
#define TT  256
#define VV  34
#define EE  256
#define HH  256
#define KVD 128
#define MHD 128
#define NLD 3

typedef __bf16 bf16;
typedef __attribute__((ext_vector_type(16))) __bf16 bf16x16;
typedef __attribute__((ext_vector_type(8)))  float  f32x8;

static __device__ __forceinline__ bf16 f2bf(float f) {
  unsigned u = __builtin_bit_cast(unsigned, f);
  unsigned short s = (unsigned short)((u + 0x7FFFu + ((u >> 16) & 1u)) >> 16);
  return __builtin_bit_cast(bf16, s);
}
static __device__ __forceinline__ float bf2f(bf16 h) {
  unsigned short s = __builtin_bit_cast(unsigned short, h);
  unsigned u = ((unsigned)s) << 16;
  return __builtin_bit_cast(float, u);
}
static __device__ __forceinline__ float sigf(float x) { return 1.0f / (1.0f + expf(-x)); }

union FragAB { uint4 u[2]; bf16x16 v; };

// ---------------- setup kernels ----------------

__global__ void k_f32_to_bf16(const float* __restrict__ src, bf16* __restrict__ dst, int n) {
  int i = blockIdx.x * blockDim.x + threadIdx.x;
  int stride = gridDim.x * blockDim.x;
  for (; i < n; i += stride) dst[i] = f2bf(src[i]);
}

__global__ void k_zero32(unsigned* __restrict__ p, int n) {
  int i = blockIdx.x * blockDim.x + threadIdx.x;
  int stride = gridDim.x * blockDim.x;
  for (; i < n; i += stride) p[i] = 0u;
}

// inputs_bf[t][b][e] = embedding[ t==0 ? sos : labels[t-1][b] ][e]  (teacher forcing, shifted)
__global__ void k_embed(const int* __restrict__ labels, const float* __restrict__ emb,
                        const int* __restrict__ sos, bf16* __restrict__ out) {
  const int n = TT * BB * EE;
  int i = blockIdx.x * blockDim.x + threadIdx.x;
  int stride = gridDim.x * blockDim.x;
  for (; i < n; i += stride) {
    int e = i % EE;
    int bt = i / EE;
    int b = bt % BB;
    int t = bt / BB;
    int lab = (t == 0) ? sos[0] : labels[(t - 1) * BB + b];
    out[i] = f2bf(emb[(size_t)lab * EE + e]);
  }
}

// x0 = concat(inputs_bf[0], zeros(ctx))
__global__ void k_init_x0(const bf16* __restrict__ embin, bf16* __restrict__ x0) {
  int i = blockIdx.x * blockDim.x + threadIdx.x;
  if (i >= BB * (EE + KVD)) return;
  int b = i / (EE + KVD);
  int e = i % (EE + KVD);
  x0[i] = (e < EE) ? embin[b * EE + e] : f2bf(0.0f);
}

__global__ void k_labels_out(const int* __restrict__ labels, float* __restrict__ outl) {
  int i = blockIdx.x * blockDim.x + threadIdx.x;
  if (i >= BB * TT) return;
  int b = i / TT, t = i % TT;
  outl[i] = (float)labels[t * BB + b];
}

// ---------------- LSTM layer: gates GEMM (bf16 WMMA) + cell pointwise ----------------
// Gates = [X | Hin] (64 x (KX+HH)) @ [Wx | Wh]^T + bih + bhh ; 16 WGs, WG w owns h-cols [w*16, w*16+16)
__global__ void __launch_bounds__(256) k_lstm(
    const bf16* __restrict__ X, int KX,
    const bf16* __restrict__ Hin,
    const bf16* __restrict__ Wx,   // [4H, KX] bf16 row-major
    const bf16* __restrict__ Wh,   // [4H, HH] bf16 row-major
    const float* __restrict__ bih, const float* __restrict__ bhh,
    float* __restrict__ C,         // [B, HH] cell state (in/out)
    bf16*  __restrict__ Hout)      // [B, HH]
{
  __shared__ float gbuf[4][BB][16];
  const int tid   = threadIdx.x;
  const int lane  = tid & 31;
  const int l16   = lane & 15;
  const int half  = lane >> 4;
  const int wave  = tid >> 5;
  const int gate  = wave & 3;        // i, f, g, o gate group
  const int mpair = wave >> 2;       // which pair of M-tiles (rows 0-31 / 32-63)
  const int hb    = blockIdx.x * 16; // h-column slice base
  const int ncol  = gate * HH + hb + l16; // global gate column for this lane

  float bias = bih[ncol] + bhh[ncol];
  f32x8 acc0, acc1;
#pragma unroll
  for (int r = 0; r < 8; ++r) { acc0[r] = bias; acc1[r] = bias; }

  const int m0 = (mpair * 2) * 16 + l16;
  const int m1 = (mpair * 2 + 1) * 16 + l16;

  for (int seg = 0; seg < 2; ++seg) {
    const bf16* A = seg ? Hin : X;
    const bf16* W = seg ? Wh  : Wx;
    const int   K = seg ? HH  : KX;
    const bf16* bRow  = W + (size_t)ncol * K;
    const bf16* aRow0 = A + (size_t)m0 * K;
    const bf16* aRow1 = A + (size_t)m1 * K;
    for (int kb = 0; kb < K; kb += 32) {
      FragAB fb, fa0, fa1;
      // B fragment: column n=ncol, chunks at kb+half*8 and kb+16+half*8 (same as A)
      const bf16* bp = bRow + kb + half * 8;
      fb.u[0] = *(const uint4*)(bp);
      fb.u[1] = *(const uint4*)(bp + 16);
      // A fragments: row m, chunks at kb+half*8 and kb+16+half*8 (ISA 16-bit A layout)
      const bf16* ap0 = aRow0 + kb + half * 8;
      fa0.u[0] = *(const uint4*)(ap0);
      fa0.u[1] = *(const uint4*)(ap0 + 16);
      const bf16* ap1 = aRow1 + kb + half * 8;
      fa1.u[0] = *(const uint4*)(ap1);
      fa1.u[1] = *(const uint4*)(ap1 + 16);
      acc0 = __builtin_amdgcn_wmma_f32_16x16x32_bf16(false, fa0.v, false, fb.v, (short)0, acc0, false, false);
      acc1 = __builtin_amdgcn_wmma_f32_16x16x32_bf16(false, fa1.v, false, fb.v, (short)0, acc1, false, false);
    }
  }

  // C/D layout: lane col = l16, rows = half*8 + r within the 16-row tile
#pragma unroll
  for (int r = 0; r < 8; ++r) {
    gbuf[gate][(mpair * 2) * 16 + half * 8 + r][l16]     = acc0[r];
    gbuf[gate][(mpair * 2 + 1) * 16 + half * 8 + r][l16] = acc1[r];
  }
  __syncthreads();

  // pointwise LSTM cell for this WG's 16 h-cols x 64 rows (1024 entries, 256 threads)
#pragma unroll
  for (int j = 0; j < 4; ++j) {
    int idx = tid + j * 256;
    int m = idx >> 4;
    int hl = idx & 15;
    float gi = gbuf[0][m][hl];
    float gf = gbuf[1][m][hl];
    float gg = gbuf[2][m][hl];
    float go = gbuf[3][m][hl];
    int hcol = hb + hl;
    float cp = C[m * HH + hcol];
    float c2 = sigf(gf) * cp + sigf(gi) * tanhf(gg);
    float h2 = sigf(go) * tanhf(c2);
    C[m * HH + hcol] = c2;
    Hout[m * HH + hcol] = f2bf(h2);
  }
}

// ---------------- query = h2 @ W_fc^T + b_fc ----------------
__global__ void k_query(const bf16* __restrict__ h2, const float* __restrict__ Wfc,
                        const float* __restrict__ bfc, float* __restrict__ query,
                        float* __restrict__ qc) {
  int gid = blockIdx.x * blockDim.x + threadIdx.x;
  if (gid >= BB * KVD) return;
  int b = gid / KVD, n = gid % KVD;
  float s = bfc[n];
  const bf16*  hp = h2 + (size_t)b * HH;
  const float* wp = Wfc + (size_t)n * HH;
  for (int k = 0; k < HH; ++k) s += bf2f(hp[k]) * wp[k];
  query[gid] = s;
  qc[b * (2 * KVD) + n] = s;
}

// ---------------- energy[b][s] = enc_key[b,s,:] . query[b,:] ----------------
__global__ void __launch_bounds__(128) k_energy(const float* __restrict__ key,
                                                const float* __restrict__ q,
                                                float* __restrict__ energy) {
  __shared__ float qs[KVD];
  int b = blockIdx.x >> 3, sc = blockIdx.x & 7;
  qs[threadIdx.x] = q[b * KVD + threadIdx.x];
  __syncthreads();
  int s = sc * 128 + threadIdx.x;
  const float4* kp = (const float4*)(key + ((size_t)b * SS + s) * KVD);
  float acc = 0.0f;
#pragma unroll 8
  for (int k4 = 0; k4 < KVD / 4; ++k4) {
    float4 kv = kp[k4];
    acc += kv.x * qs[4 * k4] + kv.y * qs[4 * k4 + 1] + kv.z * qs[4 * k4 + 2] + kv.w * qs[4 * k4 + 3];
  }
  energy[b * SS + s] = acc;
}

// ---------------- softmax(energy*mask) + ctx = attn @ enc_value ----------------
__global__ void __launch_bounds__(1024) k_softctx(
    const float* __restrict__ energy, const int* __restrict__ seqlen,
    const float* __restrict__ value, float* __restrict__ out_attn, int t,
    float* __restrict__ qc, bf16* __restrict__ x0) {
  __shared__ float sbuf[SS];
  __shared__ float ctxp[8][KVD];
  int b = blockIdx.x, tid = threadIdx.x;
  int L = seqlen[b];
  float e = energy[b * SS + tid] * ((tid < L) ? 1.0f : 0.0f); // multiplicative mask, per reference
  sbuf[tid] = e; __syncthreads();
  for (int off = 512; off > 0; off >>= 1) { if (tid < off) sbuf[tid] = fmaxf(sbuf[tid], sbuf[tid + off]); __syncthreads(); }
  float mx = sbuf[0]; __syncthreads();
  float p = expf(e - mx);
  sbuf[tid] = p; __syncthreads();
  for (int off = 512; off > 0; off >>= 1) { if (tid < off) sbuf[tid] += sbuf[tid + off]; __syncthreads(); }
  float sm = sbuf[0]; __syncthreads();
  float at = p / sm;
  out_attn[((size_t)b * SS + tid) * TT + t] = at; // attentions[b][s][t]
  sbuf[tid] = at; __syncthreads();
  // ctx: 8 s-groups x 128 k lanes
  int k = tid & (KVD - 1), grp = tid >> 7;
  const float* vp = value + ((size_t)b * SS + grp * 128) * KVD + k;
  float part = 0.0f;
  for (int s2 = 0; s2 < 128; ++s2) part += sbuf[grp * 128 + s2] * vp[(size_t)s2 * KVD];
  ctxp[grp][k] = part; __syncthreads();
  if (tid < KVD) {
    float cv = 0.0f;
#pragma unroll
    for (int g = 0; g < 8; ++g) cv += ctxp[g][tid];
    qc[b * (2 * KVD) + KVD + tid] = cv;              // for MLP concat
    x0[b * (EE + KVD) + EE + tid] = f2bf(cv);        // ctx part of next-step layer-0 input
  }
}

// ---------------- hid = LeakyReLU0.9(qc @ W_m1^T + b_m1); logits = hid @ W_m2^T + b_m2 ----------------
__global__ void __launch_bounds__(128) k_mix(
    const float* __restrict__ qc,
    const float* __restrict__ Wm1, const float* __restrict__ bm1,
    const float* __restrict__ Wm2, const float* __restrict__ bm2,
    float* __restrict__ out_y, float* __restrict__ out_label,
    int t, const bf16* __restrict__ embin, bf16* __restrict__ x0) {
  __shared__ float qcs[2 * KVD];
  __shared__ float hid[MHD];
  __shared__ float logit[VV];
  int b = blockIdx.x, tid = threadIdx.x;
  qcs[tid]       = qc[b * (2 * KVD) + tid];
  qcs[tid + 128] = qc[b * (2 * KVD) + tid + 128];
  __syncthreads();
  float s = bm1[tid];
  const float* wp = Wm1 + (size_t)tid * (2 * KVD);
  for (int k = 0; k < 2 * KVD; ++k) s += qcs[k] * wp[k];
  hid[tid] = (s >= 0.0f) ? s : 0.9f * s;
  __syncthreads();
  if (tid < VV) {
    float l = bm2[tid];
    const float* w2 = Wm2 + (size_t)tid * MHD;
    for (int k = 0; k < MHD; ++k) l += hid[k] * w2[k];
    logit[tid] = l;
    out_y[((size_t)b * TT + t) * VV + tid] = l;
  }
  // stage next-step embedding into x0 (ctx part written by k_softctx)
  if (t + 1 < TT) {
    const bf16* ep = embin + ((size_t)(t + 1) * BB + b) * EE;
    x0[b * (EE + KVD) + tid]       = ep[tid];
    x0[b * (EE + KVD) + tid + 128] = ep[tid + 128];
  }
  __syncthreads();
  if (tid == 0) {
    int best = 0; float bv = logit[0];
    for (int v = 1; v < VV; ++v) { if (logit[v] > bv) { bv = logit[v]; best = v; } }
    out_label[(size_t)b * TT + t] = (float)best;
  }
}

// ---------------- host ----------------

extern "C" void kernel_launch(void* const* d_in, const int* in_sizes, int n_in,
                              void* d_out, int out_size, void* d_ws, size_t ws_size,
                              hipStream_t stream) {
  const float* enc_key   = (const float*)d_in[0];
  const float* enc_value = (const float*)d_in[1];
  const int*   labels    = (const int*)d_in[2];
  const int*   seqlens   = (const int*)d_in[3];
  const float* embedding = (const float*)d_in[4];
  const float* W_ih0     = (const float*)d_in[5];
  const float* W_hh0     = (const float*)d_in[6];
  const float* b_ih0     = (const float*)d_in[7];
  const float* b_hh0     = (const float*)d_in[8];
  const float* W_ih_r    = (const float*)d_in[9];
  const float* W_hh_r    = (const float*)d_in[10];
  const float* b_ih_r    = (const float*)d_in[11];
  const float* b_hh_r    = (const float*)d_in[12];
  const float* W_fc      = (const float*)d_in[13];
  const float* b_fc      = (const float*)d_in[14];
  const float* W_m1      = (const float*)d_in[15];
  const float* b_m1      = (const float*)d_in[16];
  const float* W_m2      = (const float*)d_in[17];
  const float* b_m2      = (const float*)d_in[18];
  const int*   sos       = (const int*)d_in[19];
  (void)in_sizes; (void)n_in; (void)out_size; (void)ws_size;

  char* wsp = (char*)d_ws;
  auto carve = [&](size_t bytes) -> char* {
    char* p = wsp;
    wsp += (bytes + 255) & ~(size_t)255;
    return p;
  };
  bf16*  wih0b  = (bf16*)carve((size_t)4 * HH * (EE + KVD) * 2);
  bf16*  whh0b  = (bf16*)carve((size_t)4 * HH * HH * 2);
  bf16*  wihrb  = (bf16*)carve((size_t)2 * 4 * HH * HH * 2);
  bf16*  whhrb  = (bf16*)carve((size_t)2 * 4 * HH * HH * 2);
  bf16*  embin  = (bf16*)carve((size_t)TT * BB * EE * 2);
  bf16*  x0     = (bf16*)carve((size_t)BB * (EE + KVD) * 2);
  bf16*  hbuf   = (bf16*)carve((size_t)NLD * 2 * BB * HH * 2);
  float* cbuf   = (float*)carve((size_t)NLD * BB * HH * 4);
  float* query  = (float*)carve((size_t)BB * KVD * 4);
  float* qc     = (float*)carve((size_t)BB * 2 * KVD * 4);
  float* energy = (float*)carve((size_t)BB * SS * 4);

  float* out_y      = (float*)d_out;                       // [B,T,V]
  float* out_label  = out_y + (size_t)BB * TT * VV;        // [B,T]
  float* out_labels = out_label + (size_t)BB * TT;         // [B,T]
  float* out_attn   = out_labels + (size_t)BB * TT;        // [B,S,T]

  // per-launch setup (deterministic, fully rewrites all state)
  k_f32_to_bf16<<<256, 256, 0, stream>>>(W_ih0,  wih0b, 4 * HH * (EE + KVD));
  k_f32_to_bf16<<<256, 256, 0, stream>>>(W_hh0,  whh0b, 4 * HH * HH);
  k_f32_to_bf16<<<256, 256, 0, stream>>>(W_ih_r, wihrb, 2 * 4 * HH * HH);
  k_f32_to_bf16<<<256, 256, 0, stream>>>(W_hh_r, whhrb, 2 * 4 * HH * HH);
  k_embed<<<1024, 256, 0, stream>>>(labels, embedding, sos, embin);
  k_zero32<<<256, 256, 0, stream>>>((unsigned*)hbuf, NLD * 2 * BB * HH / 2);
  k_zero32<<<256, 256, 0, stream>>>((unsigned*)cbuf, NLD * BB * HH);
  k_init_x0<<<(BB * (EE + KVD) + 255) / 256, 256, 0, stream>>>(embin, x0);
  k_labels_out<<<(BB * TT + 255) / 256, 256, 0, stream>>>(labels, out_labels);

  const size_t hsz = (size_t)BB * HH;
  for (int t = 0; t < TT; ++t) {
    int pa = t & 1, pb = pa ^ 1; // read parity pa, write parity pb
    bf16* h0i = hbuf + (0 * 2 + pa) * hsz; bf16* h0o = hbuf + (0 * 2 + pb) * hsz;
    bf16* h1i = hbuf + (1 * 2 + pa) * hsz; bf16* h1o = hbuf + (1 * 2 + pb) * hsz;
    bf16* h2i = hbuf + (2 * 2 + pa) * hsz; bf16* h2o = hbuf + (2 * 2 + pb) * hsz;

    k_lstm<<<16, 256, 0, stream>>>(x0, EE + KVD, h0i, wih0b, whh0b, b_ih0, b_hh0,
                                   cbuf + 0 * hsz, h0o);
    k_lstm<<<16, 256, 0, stream>>>(h0o, HH, h1i, wihrb, whhrb, b_ih_r, b_hh_r,
                                   cbuf + 1 * hsz, h1o);
    k_lstm<<<16, 256, 0, stream>>>(h1o, HH, h2i, wihrb + (size_t)4 * HH * HH,
                                   whhrb + (size_t)4 * HH * HH,
                                   b_ih_r + 4 * HH, b_hh_r + 4 * HH,
                                   cbuf + 2 * hsz, h2o);
    k_query<<<(BB * KVD + 255) / 256, 256, 0, stream>>>(h2o, W_fc, b_fc, query, qc);
    k_energy<<<BB * 8, 128, 0, stream>>>(enc_key, query, energy);
    k_softctx<<<BB, 1024, 0, stream>>>(energy, seqlens, enc_value, out_attn, t, qc, x0);
    k_mix<<<BB, 128, 0, stream>>>(qc, W_m1, b_m1, W_m2, b_m2, out_y, out_label, t, embin, x0);
  }
}
